// CorticalRNN_27479200759885
// MI455X (gfx1250) — matrix operand order repaired and checked by reading, not compiled, not run on hardware
//
#include <hip/hip_runtime.h>
#include <math.h>

// ---------------------------------------------------------------------------
// CorticalRNN on MI455X (gfx1250, wave32, WMMA + TDM, persistent scan).
//   Phase 1: x_proj = inputs @ W_ih^T + b  (parallel GEMM, LDS-staged A panel)
//   Phase 2: ONE persistent kernel runs all 512 timesteps; 256 resident waves,
//            device-wide sense-reversal barrier between dependent GEMM phases.
//            A panels (f_prev / readout, 16x1024 bf16) are staged into LDS by
//            the Tensor Data Mover (TENSOR_LOAD_TO_LDS + s_wait_tensorcnt),
//            with TDM-inserted 16B padding per 512-element half-row so the
//            ds_load_b128 fragment reads are bank-conflict-reduced.
// Per step:
//   A: recurrent = f_prev@W_hh^T + b_hh + alpha*h + x_proj[t] -> h_tmp  (waves 0-127)
//      readout   = f_prev@W_rd^T + b_rd -> rd (bf16)                    (waves 128-255)
//   B: h_new = h_tmp + rd@W_zh^T + b_zh ; out = tanh(h_new) ; f_prev = bf16(out)
// ---------------------------------------------------------------------------

#define ALPHA_C   0.1f
#define HIDDEN_C  1024
#define INPUT_C   512
#define BATCH_C   32
#define TSTEPS_C  512
#define SCAN_BLOCKS 64u

typedef __bf16 bf16_t;
typedef __attribute__((ext_vector_type(16))) __bf16 v16bf;
typedef __attribute__((ext_vector_type(8)))  float  v8f;
typedef __attribute__((ext_vector_type(4)))  unsigned int v4u;
typedef __attribute__((ext_vector_type(8)))  int v8i;
typedef __attribute__((ext_vector_type(4)))  int v4i;

// x_proj kernel LDS panel: manual staging, +8 bf16 pad per row (K=512)
#define LDS_LDR_I (INPUT_C + 8)

// Scan kernel LDS panel (TDM-padded): 32 sub-rows (16 rows x 2 halves) of
// 512 bf16 data + 8 bf16 (16 B = 4 DWORD) TDM pad  => 520 elem / 1040 B stride.
#define SUBROW_ELEMS 520
#define SCAN_PANEL_ELEMS (32 * SUBROW_ELEMS)   // 33,280 B

union frag32 { uint4 q[2]; v16bf v; };

// ---- WMMA operand loaders (wave32 layouts, CDNA5 ISA 7.12.2) --------------

// A tile 16x32 (MxK) bf16 from plain padded LDS panel (ldr elems / row):
__device__ __forceinline__ v16bf load_a_lds(const bf16_t* __restrict__ P,
                                            int ldr, int k0, int lane) {
  const int row = lane & 15;
  const int kb  = (lane >> 4) * 8;
  const bf16_t* p = P + row * ldr + k0 + kb;
  frag32 u;
  u.q[0] = *(const uint4*)p;          // K = k0+kb+0..7
  u.q[1] = *(const uint4*)(p + 16);   // K = k0+kb+16..23
  return u.v;
}

// A tile from the TDM-padded panel: element (row,k) lives in sub-row
// (row*2 + k/512) at offset (k%512); sub-row stride 520 elems (1040 B).
// Fragments never straddle a 512 boundary (k0 % 32 == 0, span < 32).
__device__ __forceinline__ v16bf load_a_lds_tdm(const bf16_t* __restrict__ P,
                                                int k0, int lane) {
  const int sub = ((lane & 15) << 1) + (k0 >> 9);
  const int kk  = (k0 & 511) + ((lane >> 4) << 3);
  const bf16_t* p = P + sub * SUBROW_ELEMS + kk;
  frag32 u;
  u.q[0] = *(const uint4*)p;
  u.q[1] = *(const uint4*)(p + 16);
  return u.v;
}

// B tile 32x16 (KxN) bf16, y = f @ W^T so B[k][n] = W[n][k] (W row-major):
__device__ __forceinline__ v16bf load_b_tile(const bf16_t* __restrict__ W,
                                             int ldw, int n0, int k0, int lane) {
  const int col = n0 + (lane & 15);
  const int kb  = (lane >> 4) * 8;
  const bf16_t* p = W + (size_t)col * ldw + k0 + kb;
  frag32 u;
  u.q[0] = *(const uint4*)p;
  u.q[1] = *(const uint4*)(p + 16);
  return u.v;
}

__device__ __forceinline__ v8f wmma_bf16(v16bf a, v16bf b, v8f c) {
  return __builtin_amdgcn_wmma_f32_16x16x32_bf16(
      false, a, false, b, (short)0, c, false, false);
}

// ---- Tensor Data Mover: 16x1024 bf16 panel -> LDS (padded) ----------------
// D# per CDNA5 ISA ch.8: group0 = {count/lds_addr/global_addr/type},
// group1 = {data_size, pad ctrl, tensor/tile dims, dim0 stride}.
__device__ __forceinline__ void tdm_load_panel(unsigned lds_off,
                                               const bf16_t* __restrict__ gsrc) {
  const unsigned long long ga = (unsigned long long)(uintptr_t)gsrc;
  v4u g0;
  g0.x = 0x1u;                                           // count=1, user desc
  g0.y = lds_off;                                        // lds_addr (bytes)
  g0.z = (unsigned)(ga & 0xffffffffull);                 // global_addr[31:0]
  g0.w = (unsigned)((ga >> 32) & 0x01ffffffull)          // global_addr[56:32]
       | (2u << 30);                                     // type = 2 ("image")
  v8i g1;
  g1[0] = (1 << 16)      // data_size = 1 -> 2 bytes
        | (1 << 20)      // pad_enable
        | (7 << 22)      // pad_interval = 7 -> every 256 DWORDs (1024 B)
        | (3 << 25);     // pad_amount  = 3 -> 4 DWORDs (16 B)
  g1[1] = (int)(((unsigned)HIDDEN_C & 0xffffu) << 16);   // tensor_dim0 lo
  g1[2] = (int)(16u << 16);                              // dim0 hi=0; tensor_dim1=16 lo
  g1[3] = (int)(((unsigned)HIDDEN_C & 0xffffu) << 16);   // dim1 hi=0; tile_dim0=1024
  g1[4] = 16;                                            // tile_dim1=16; tile_dim2=0
  g1[5] = HIDDEN_C;                                      // tensor_dim0_stride lo
  g1[6] = 0;                                             // stride hi; dim1_stride lo
  g1[7] = 0;
  v4i z4 = {0, 0, 0, 0};
#if __clang_major__ >= 23
  v8i z8 = {0, 0, 0, 0, 0, 0, 0, 0};
  __builtin_amdgcn_tensor_load_to_lds(g0, g1, z4, z4, z8, 0);
#else
  __builtin_amdgcn_tensor_load_to_lds(g0, g1, z4, z4, 0);
#endif
}

// Manual LDS staging (x_proj kernel): 16 rows x K bf16, padded rows.
template <int K, int LDR>
__device__ __forceinline__ void stage_panel(bf16_t* __restrict__ lds,
                                            const bf16_t* __restrict__ src,
                                            int m0, int nthreads) {
  const int chunks_per_row = K / 8;
  const int total = 16 * chunks_per_row;
  for (int i = threadIdx.x; i < total; i += nthreads) {
    const int r = i / chunks_per_row;
    const int c = i % chunks_per_row;
    *(uint4*)(lds + r * LDR + c * 8) =
        *(const uint4*)(src + (size_t)(m0 + r) * K + c * 8);
  }
}

// ---- device-wide sense-reversal barrier (all SCAN_BLOCKS resident) --------
__device__ __forceinline__ void grid_sync(unsigned* cnt, unsigned* sense) {
  __syncthreads();
  if (threadIdx.x == 0) {
    volatile unsigned* vs = sense;
    const unsigned s = vs[0];
    __threadfence();                                  // release my stores
    if (atomicAdd(cnt, 1u) == SCAN_BLOCKS - 1u) {
      *cnt = 0u;
      __threadfence();
      atomicAdd(sense, 1u);                           // open the barrier
    } else {
      while (vs[0] == s) __builtin_amdgcn_s_sleep(2);
      __threadfence();                                // acquire others' stores
    }
  }
  __syncthreads();
}

// ---- utility kernels ------------------------------------------------------

__global__ void cvt_f32_to_bf16(const float* __restrict__ src,
                                bf16_t* __restrict__ dst, int n) {
  int i = blockIdx.x * blockDim.x + threadIdx.x;
  const int stride = gridDim.x * blockDim.x;
  for (; i < n; i += stride) dst[i] = (bf16_t)src[i];
}

__global__ void init_state(float* __restrict__ h, bf16_t* __restrict__ f,
                           int n, unsigned* __restrict__ sync) {
  int i = blockIdx.x * blockDim.x + threadIdx.x;
  if (i < n) { h[i] = 0.0f; f[i] = (bf16_t)0.0f; }   // tanh(0) == 0
  if (i < 2) sync[i] = 0u;                            // barrier count+sense
}

// ---- phase 1: x_proj GEMM  (M = B*T = 16384, N = 1024, K = 512) -----------

__global__ __launch_bounds__(256)
void xproj_gemm(const bf16_t* __restrict__ Xbf,    // [B*T][512]
                const bf16_t* __restrict__ Wih,    // [1024][512]
                const float*  __restrict__ bih,    // [1024]
                float* __restrict__ xproj) {       // [B*T][1024]
  __shared__ bf16_t Ash[16 * LDS_LDR_I];
  const int lane = threadIdx.x & 31;
  const int wave = blockIdx.x * 8 + (threadIdx.x >> 5);
  const int m0 = (wave >> 6) * 16;                 // uniform per block
  const int n0 = (wave & 63) * 16;

  stage_panel<INPUT_C, LDS_LDR_I>(Ash, Xbf, m0, 256);
  __syncthreads();

  v8f acc = {};
#pragma unroll 4
  for (int k0 = 0; k0 < INPUT_C; k0 += 32) {
    v16bf a = load_a_lds(Ash, LDS_LDR_I, k0, lane);
    v16bf b = load_b_tile(Wih, INPUT_C, n0, k0, lane);
    acc = wmma_bf16(a, b, acc);
  }

  const int lrow = (lane >> 4) * 8;
  const int lcol = lane & 15;
  const float bias = bih[n0 + lcol];
#pragma unroll
  for (int r = 0; r < 8; ++r)
    xproj[(size_t)(m0 + lrow + r) * HIDDEN_C + n0 + lcol] = acc[r] + bias;
}

// ---- phase 2: persistent scan kernel --------------------------------------
// 64 blocks x 128 threads = 256 waves. wid = 0..255:
//   g = wid>>7 (0: W_hh recurrent / phase-B owner, 1: W_rd readout)
//   tile = wid&127 -> m0 = (tile>>6)*16, n0 = (tile&63)*16
// g and m0 are uniform within a block (4 consecutive wids per block).

__global__ __launch_bounds__(128)
void scan_persistent(const bf16_t* __restrict__ Whh, const float* __restrict__ bhh,
                     const bf16_t* __restrict__ Wrd, const float* __restrict__ brd,
                     const bf16_t* __restrict__ Wzh, const float* __restrict__ bzh,
                     const float*  __restrict__ xproj,   // [B*T][1024]
                     float*  __restrict__ h,             // [32][1024]
                     bf16_t* __restrict__ fprev,         // [32][1024]
                     float*  __restrict__ htmp,          // [32][1024]
                     bf16_t* __restrict__ rdbf,          // [32][1024]
                     float*  __restrict__ out,           // [B][T][1024]
                     unsigned* __restrict__ sync) {      // [0]=count [1]=sense
  __shared__ bf16_t Ash[SCAN_PANEL_ELEMS];               // 33,280 B
  const int lane = threadIdx.x & 31;
  const int wid  = blockIdx.x * 4 + (threadIdx.x >> 5);  // 0..255
  const int g    = wid >> 7;
  const int tile = wid & 127;
  const int m0   = (tile >> 6) * 16;
  const int n0   = (tile & 63) * 16;
  const int lrow = (lane >> 4) * 8;
  const int col  = n0 + (lane & 15);
  const bf16_t* WA = g ? Wrd : Whh;
  unsigned* cnt   = sync;
  unsigned* sense = sync + 1;

  // Low 32 bits of the LDS-aperture generic address == LDS byte offset
  // (ISA 10.2).
  const unsigned lds_off = (unsigned)(uintptr_t)(&Ash[0]);
  // Loop-invariant TDM source panels for this block:
  const bf16_t* panelF = fprev + (size_t)m0 * HIDDEN_C;  // phase A
  const bf16_t* panelR = rdbf  + (size_t)m0 * HIDDEN_C;  // phase B

  for (int t = 0; t < TSTEPS_C; ++t) {
    // ---------------- Phase A: recurrent (g=0) & readout (g=1) ------------
    if (threadIdx.x < 32) {            // wave 0 drives the TDM (per-wave cnt)
      tdm_load_panel(lds_off, panelF);
      __builtin_amdgcn_s_wait_tensorcnt(0);
    }
    __syncthreads();

    v8f acc = {};
#pragma unroll 4
    for (int k0 = 0; k0 < HIDDEN_C; k0 += 32) {
      v16bf a = load_a_lds_tdm(Ash, k0, lane);
      v16bf b = load_b_tile(WA, HIDDEN_C, n0, k0, lane);
      acc = wmma_bf16(a, b, acc);
    }

    if (g == 0) {
      const float bias = bhh[col];
#pragma unroll
      for (int r = 0; r < 8; ++r) {
        const int row = m0 + lrow + r;                   // batch index
        const size_t hi = (size_t)row * HIDDEN_C + col;
        const float xp = xproj[((size_t)row * TSTEPS_C + t) * HIDDEN_C + col];
        htmp[hi] = acc[r] + bias + ALPHA_C * h[hi] + xp;
      }
    } else {
      const float bias = brd[col];
#pragma unroll
      for (int r = 0; r < 8; ++r) {
        const size_t hi = (size_t)(m0 + lrow + r) * HIDDEN_C + col;
        rdbf[hi] = (bf16_t)(acc[r] + bias);
      }
    }
    grid_sync(cnt, sense);

    // ---------------- Phase B: correction + state update (g=0 blocks) -----
    if (g == 0) {                                        // block-uniform
      if (threadIdx.x < 32) {
        tdm_load_panel(lds_off, panelR);
        __builtin_amdgcn_s_wait_tensorcnt(0);
      }
      __syncthreads();

      v8f acc2 = {};
#pragma unroll 4
      for (int k0 = 0; k0 < HIDDEN_C; k0 += 32) {
        v16bf a = load_a_lds_tdm(Ash, k0, lane);
        v16bf b = load_b_tile(Wzh, HIDDEN_C, n0, k0, lane);
        acc2 = wmma_bf16(a, b, acc2);
      }

      const float bias = bzh[col];
#pragma unroll
      for (int r = 0; r < 8; ++r) {
        const int row = m0 + lrow + r;
        const size_t hi = (size_t)row * HIDDEN_C + col;
        const float hnew = htmp[hi] + acc2[r] + bias;
        h[hi] = hnew;
        const float f = tanhf(hnew);
        fprev[hi] = (bf16_t)f;
        out[((size_t)row * TSTEPS_C + t) * HIDDEN_C + col] = f;
      }
    }
    grid_sync(cnt, sense);
  }
}

// ---- host side ------------------------------------------------------------

extern "C" void kernel_launch(void* const* d_in, const int* in_sizes, int n_in,
                              void* d_out, int out_size, void* d_ws, size_t ws_size,
                              hipStream_t stream) {
  const float* inputs = (const float*)d_in[0];  // [32,512,512]
  const float* W_hh_w = (const float*)d_in[1];  // [1024,1024]
  const float* W_hh_b = (const float*)d_in[2];  // [1024]
  const float* W_ih_w = (const float*)d_in[3];  // [1024,512]
  const float* W_ih_b = (const float*)d_in[4];  // [1024]
  const float* W_rd_w = (const float*)d_in[5];  // [1024,1024]
  const float* W_rd_b = (const float*)d_in[6];  // [1024]
  const float* W_zh_w = (const float*)d_in[7];  // [1024,1024]
  const float* W_zh_b = (const float*)d_in[8];  // [1024]
  float* out = (float*)d_out;                   // [32,512,1024]

  uint8_t* base = (uint8_t*)d_ws;
  size_t off = 0;
  auto carve = [&](size_t bytes) -> void* {
    void* p = base + off;
    off = (off + bytes + 255) & ~(size_t)255;
    return p;
  };
  const int NW  = HIDDEN_C * HIDDEN_C;
  const int NWI = HIDDEN_C * INPUT_C;
  const int NX  = BATCH_C * TSTEPS_C * INPUT_C;
  const int NXP = BATCH_C * TSTEPS_C * HIDDEN_C;
  const int NH  = BATCH_C * HIDDEN_C;

  bf16_t*   Wih_bf = (bf16_t*)carve((size_t)NWI * 2);
  bf16_t*   Whh_bf = (bf16_t*)carve((size_t)NW  * 2);
  bf16_t*   Wrd_bf = (bf16_t*)carve((size_t)NW  * 2);
  bf16_t*   Wzh_bf = (bf16_t*)carve((size_t)NW  * 2);
  bf16_t*   Xbf    = (bf16_t*)carve((size_t)NX  * 2);
  float*    xproj  = (float*) carve((size_t)NXP * 4);
  float*    h      = (float*) carve((size_t)NH  * 4);
  float*    htmp   = (float*) carve((size_t)NH  * 4);
  bf16_t*   fprev  = (bf16_t*)carve((size_t)NH  * 2);
  bf16_t*   rdbf   = (bf16_t*)carve((size_t)NH  * 2);
  unsigned* sync   = (unsigned*)carve(256);
  (void)ws_size; (void)in_sizes; (void)n_in; (void)out_size;

  // 1) bf16 conversions + state/barrier init
  cvt_f32_to_bf16<<<2048, 256, 0, stream>>>(W_ih_w, Wih_bf, NWI);
  cvt_f32_to_bf16<<<2048, 256, 0, stream>>>(W_hh_w, Whh_bf, NW);
  cvt_f32_to_bf16<<<2048, 256, 0, stream>>>(W_rd_w, Wrd_bf, NW);
  cvt_f32_to_bf16<<<2048, 256, 0, stream>>>(W_zh_w, Wzh_bf, NW);
  cvt_f32_to_bf16<<<4096, 256, 0, stream>>>(inputs, Xbf, NX);
  init_state<<<(NH + 255) / 256, 256, 0, stream>>>(h, fprev, NH, sync);

  // 2) x_proj: 1024 M-tiles x 64 N-tiles, 8 waves/block sharing an LDS panel
  xproj_gemm<<<8192, 256, 0, stream>>>(Xbf, Wih_bf, W_ih_b, xproj);

  // 3) the whole 512-step scan in ONE persistent kernel (2 grid syncs/step),
  //    A panels staged by the Tensor Data Mover
  scan_persistent<<<SCAN_BLOCKS, 128, 0, stream>>>(
      Whh_bf, W_hh_b, Wrd_bf, W_rd_b, Wzh_bf, W_zh_b,
      xproj, h, fprev, htmp, rdbf, out, sync);
}
